// CustomLSTM_38010460569797
// MI455X (gfx1250) — matrix operand order repaired and checked by reading, not compiled, not run on hardware
//
#include <hip/hip_runtime.h>

// ---------------------------------------------------------------------------
// 2-layer LSTM for MI455X (gfx1250, wave32, WMMA bf16 16x16x32, f32 accum)
// T=512, B=64, I=H=1024.  d_out = [ out1 (512*64*1024) | h_f (2*64*1024) |
//                                   c_f (2*64*1024) ]  (fp32)
// ---------------------------------------------------------------------------

typedef __bf16 bf16;
typedef __bf16 v16bf __attribute__((ext_vector_type(16)));
typedef __bf16 v8bf  __attribute__((ext_vector_type(8)));
typedef float  v8f   __attribute__((ext_vector_type(8)));

#define HID    1024
#define B4H    4096
#define KCAT   2048          // concat [x | h] K dimension
#define BATCH  64
#define TSTEPS 512
#define STEPEL (BATCH * HID) // 65536 elements per (t) slab

// ---------------------------------------------------------------------------
// Pack weights:  Wpk[((kt*4096 + N)*2 + half)*16 + j] = bf16( Wsrc[k][N] )
// where k = kt*32 + half*16 + j and Wsrc = [U^T ; V^T] (K=2048 rows, N=4096).
// This is the per-lane B-fragment image in this code's contiguous k order (k = 16*half + j),
// so the step kernel loads each fragment as one aligned 32-byte chunk.
// ---------------------------------------------------------------------------
__global__ __launch_bounds__(256) void prep_weights(
    const float* __restrict__ U,   // [4096, 1024] row-major (gate-major)
    const float* __restrict__ V,   // [4096, 1024]
    unsigned short* __restrict__ Wpk_u16)
{
    bf16* Wpk = (bf16*)Wpk_u16;
    const size_t idx = (size_t)blockIdx.x * blockDim.x + threadIdx.x; // < 2048*4096
    const int j    = (int)(idx & 15);
    const int half = (int)((idx >> 4) & 1);
    const int N    = (int)((idx >> 5) & 4095);
    const int kt   = (int)(idx >> 17);
    const int k    = (kt << 5) + (half << 4) + j;
    const float v = (k < HID) ? U[(size_t)N * HID + k]
                              : V[(size_t)N * HID + (k - HID)];
    Wpk[idx] = (bf16)v;
}

__global__ __launch_bounds__(256) void zero_u32(unsigned* __restrict__ p, int n)
{
    const int i = blockIdx.x * blockDim.x + threadIdx.x;
    if (i < n) p[i] = 0u;
}

// ---------------------------------------------------------------------------
// Fragment loaders (contiguous k order: lane half h holds k = 16h + 0..15).
// ---------------------------------------------------------------------------
__device__ __forceinline__ v16bf ldA_f32(const float* p)
{
    const float4* q = (const float4*)p;          // 64B-aligned (ko mult of 16)
    const float4 f0 = q[0], f1 = q[1], f2 = q[2], f3 = q[3];
    v16bf a;
    a[0]  = (bf16)f0.x; a[1]  = (bf16)f0.y; a[2]  = (bf16)f0.z; a[3]  = (bf16)f0.w;
    a[4]  = (bf16)f1.x; a[5]  = (bf16)f1.y; a[6]  = (bf16)f1.z; a[7]  = (bf16)f1.w;
    a[8]  = (bf16)f2.x; a[9]  = (bf16)f2.y; a[10] = (bf16)f2.z; a[11] = (bf16)f2.w;
    a[12] = (bf16)f3.x; a[13] = (bf16)f3.y; a[14] = (bf16)f3.z; a[15] = (bf16)f3.w;
    return a;
}

__device__ __forceinline__ v16bf ldA_bf16(const bf16* p)
{
    const v8bf* q = (const v8bf*)p;              // 16B-aligned
    const v8bf lo = q[0], hi = q[1];             // K = +0..7 and +8..15
    v16bf a;
    #pragma unroll
    for (int j = 0; j < 8; ++j) { a[j] = lo[j]; a[8 + j] = hi[j]; }
    return a;
}

__device__ __forceinline__ v16bf ldB(const bf16* p) { return *(const v16bf*)p; }

// ---------------------------------------------------------------------------
// One fused LSTM step (template XF32: 1 = x input is f32, 0 = bf16).
//   gates[64,4096] = [x_t | h_{t-1}] @ Wcat  via bf16 WMMA, f32 accum,
//   then the LSTM cell entirely in-register.
// Grid: 32 blocks x 256 threads = 256 waves; wave gw owns rows m0..m0+15 and
// hidden columns n0..n0+15 in ALL FOUR gate blocks (i,f,g,o) so the cell
// needs no cross-wave exchange.
// Depth-2 software pipeline: stage s = kt&1 computes while being refilled
// with fragments for kt+2.
// ---------------------------------------------------------------------------
template <int XF32>
__global__ __launch_bounds__(256) void lstm_step(
    const void* __restrict__ xin,                 // f32 or bf16 [64,1024]
    const unsigned short* __restrict__ hprev_u16, // bf16 [64,1024]
    const unsigned short* __restrict__ Wpk_u16,   // packed bf16 weights
    const float* __restrict__ bih,
    const float* __restrict__ bhh,
    float* __restrict__ cst,                      // f32 [64,1024] in/out
    unsigned short* __restrict__ hout_u16,        // bf16 [64,1024] out
    float* __restrict__ out_f32,                  // optional f32 h out
    float* __restrict__ hf_out,                   // optional final-h f32 out
    float* __restrict__ cf_out)                   // optional final-c f32 out
{
    const bf16* hprev = (const bf16*)hprev_u16;
    const bf16* Wpk   = (const bf16*)Wpk_u16;
    bf16*       hout  = (bf16*)hout_u16;
    const float* ax   = (const float*)xin;
    const bf16*  ab   = (const bf16*)xin;

    const int tid   = threadIdx.x;
    const int wave  = tid >> 5;
    const int lane  = tid & 31;
    const int gw    = blockIdx.x * 8 + wave;   // 0..255
    const int mtile = gw & 3;                  // row tile 0..3
    const int ng    = gw >> 2;                 // col group 0..63
    const int m0    = mtile << 4;
    const int n0    = ng << 4;
    const int l     = lane & 15;
    const int lhalf = lane >> 4;
    const int ko    = lhalf << 4;              // A K-offset (0 or 16)
    const size_t rowoff = (size_t)(m0 + l) * HID;

    // B fragment base for this lane: ((kt*4096 + g*1024 + n0+l)*2 + lhalf)*16
    const bf16* Bbase = Wpk + (size_t)(n0 + l) * 32 + (size_t)lhalf * 16;
#define BOFF(kt, g) ((size_t)(kt) * 131072 + (size_t)(g) * 32768)
#define LDA_X(kt) (XF32 ? ldA_f32(ax + rowoff + (size_t)(kt) * 32 + ko) \
                        : ldA_bf16(ab + rowoff + (size_t)(kt) * 32 + ko))
#define LDA_H(kt) ldA_bf16(hprev + rowoff + (size_t)((kt) - 32) * 32 + ko)
#define LDB4(s, kt)                              \
    Bf[s][0] = ldB(Bbase + BOFF(kt, 0));         \
    Bf[s][1] = ldB(Bbase + BOFF(kt, 1));         \
    Bf[s][2] = ldB(Bbase + BOFF(kt, 2));         \
    Bf[s][3] = ldB(Bbase + BOFF(kt, 3));
#define STEP(s)                                                                                   \
    acc0 = __builtin_amdgcn_wmma_f32_16x16x32_bf16(false, A[s], false, Bf[s][0], (short)0, acc0, false, false); \
    acc1 = __builtin_amdgcn_wmma_f32_16x16x32_bf16(false, A[s], false, Bf[s][1], (short)0, acc1, false, false); \
    acc2 = __builtin_amdgcn_wmma_f32_16x16x32_bf16(false, A[s], false, Bf[s][2], (short)0, acc2, false, false); \
    acc3 = __builtin_amdgcn_wmma_f32_16x16x32_bf16(false, A[s], false, Bf[s][3], (short)0, acc3, false, false);

    v8f acc0 = {}, acc1 = {}, acc2 = {}, acc3 = {};
    v16bf A[2];
    v16bf Bf[2][4];

    // prologue: stages for kt=0,1 (X phase)
    A[0] = LDA_X(0); LDB4(0, 0);
    A[1] = LDA_X(1); LDB4(1, 1);

    // phase 1: kt = 0..29, refill kt+2 from X
    #pragma unroll 2
    for (int kt = 0; kt < 30; ++kt) {
        const int s = kt & 1;
        STEP(s);
        A[s] = LDA_X(kt + 2);
        LDB4(s, kt + 2);
    }
    // boundary: kt=30,31 compute X, refill kt+2 = 32,33 from H
    { STEP(0); A[0] = LDA_H(32); LDB4(0, 32); }
    { STEP(1); A[1] = LDA_H(33); LDB4(1, 33); }

    // phase 2: kt = 32..61, refill kt+2 from H
    #pragma unroll 2
    for (int kt = 32; kt < 62; ++kt) {
        const int s = kt & 1;
        STEP(s);
        A[s] = LDA_H(kt + 2);
        LDB4(s, kt + 2);
    }
    // epilogue: kt = 62, 63
    { STEP(0); }
    { STEP(1); }

#undef BOFF
#undef LDA_X
#undef LDA_H
#undef LDB4
#undef STEP

    // ---- LSTM cell: lane owns column col for 8 rows (C/D f32 layout) ----
    const int col = n0 + l;
    const float bi  = bih[0 * HID + col] + bhh[0 * HID + col];
    const float bf_ = bih[1 * HID + col] + bhh[1 * HID + col];
    const float bg  = bih[2 * HID + col] + bhh[2 * HID + col];
    const float bo  = bih[3 * HID + col] + bhh[3 * HID + col];

    #pragma unroll
    for (int rr = 0; rr < 8; ++rr) {
        const int row = m0 + rr + (lhalf << 3);
        const size_t idx = (size_t)row * HID + col;
        const float iv = 1.0f / (1.0f + __expf(-(acc0[rr] + bi)));
        const float fv = 1.0f / (1.0f + __expf(-(acc1[rr] + bf_)));
        const float gv = tanhf(acc2[rr] + bg);
        const float ov = 1.0f / (1.0f + __expf(-(acc3[rr] + bo)));
        const float cn = fv * cst[idx] + iv * gv;
        const float hv = ov * tanhf(cn);
        cst[idx]  = cn;
        hout[idx] = (bf16)hv;
        if (out_f32) out_f32[idx] = hv;
        if (hf_out)  hf_out[idx]  = hv;
        if (cf_out)  cf_out[idx]  = cn;
    }
}

// ---------------------------------------------------------------------------
// Host orchestration.  Workspace (~34.6 MB):
//   [0,16MB) W0 packed | [16,32MB) W1 packed | h0 pp | h1 pp | c0 | c1
// ---------------------------------------------------------------------------
extern "C" void kernel_launch(void* const* d_in, const int* in_sizes, int n_in,
                              void* d_out, int out_size, void* d_ws, size_t ws_size,
                              hipStream_t stream)
{
    const float* x    = (const float*)d_in[0];
    const float* U0   = (const float*)d_in[1];
    const float* V0   = (const float*)d_in[2];
    const float* bih0 = (const float*)d_in[3];
    const float* bhh0 = (const float*)d_in[4];
    const float* U1   = (const float*)d_in[5];
    const float* V1   = (const float*)d_in[6];
    const float* bih1 = (const float*)d_in[7];
    const float* bhh1 = (const float*)d_in[8];
    float* out = (float*)d_out;

    char* ws = (char*)d_ws;
    const size_t WBYTES = (size_t)KCAT * B4H * 2;           // 16 MB per layer
    unsigned short* W0 = (unsigned short*)(ws);
    unsigned short* W1 = (unsigned short*)(ws + WBYTES);
    unsigned short* H0 = (unsigned short*)(ws + 2 * WBYTES);           // 2*65536 bf16
    unsigned short* H1 = (unsigned short*)(ws + 2 * WBYTES + 262144);
    float* C0 = (float*)(ws + 2 * WBYTES + 2 * 262144);
    float* C1 = (float*)(ws + 2 * WBYTES + 3 * 262144);

    // 1) pack + bf16-convert weights (lives in the 192MB L2 across all steps)
    const int prepThreads = 256;
    const int prepBlocks  = (KCAT * B4H) / prepThreads;     // 32768
    prep_weights<<<prepBlocks, prepThreads, 0, stream>>>(U0, V0, W0);
    prep_weights<<<prepBlocks, prepThreads, 0, stream>>>(U1, V1, W1);

    // 2) zero h ping-pongs and c states (1 MB)
    zero_u32<<<1024, 256, 0, stream>>>((unsigned*)(ws + 2 * WBYTES), 262144);

    const size_t OUT1_ELEMS = (size_t)TSTEPS * STEPEL;      // 33554432
    float* hf = out + OUT1_ELEMS;                           // [2,64,1024]
    float* cf = out + OUT1_ELEMS + 2 * STEPEL;              // [2,64,1024]

    // 3) sequential scan: layer0 step t, then layer1 step t (consumes h0[t])
    for (int t = 0; t < TSTEPS; ++t) {
        const int pp  = t & 1;        // write buffer
        const int pn  = pp ^ 1;       // read buffer (t-1 output; zeroed at t=0)
        const bool last = (t == TSTEPS - 1);

        const float*    x_t    = x + (size_t)t * STEPEL;
        unsigned short* h0prev = H0 + (size_t)pn * STEPEL;
        unsigned short* h0out  = H0 + (size_t)pp * STEPEL;
        unsigned short* h1prev = H1 + (size_t)pn * STEPEL;
        unsigned short* h1out  = H1 + (size_t)pp * STEPEL;

        lstm_step<1><<<32, 256, 0, stream>>>(
            (const void*)x_t, h0prev, W0, bih0, bhh0,
            C0, h0out,
            /*out_f32=*/nullptr,
            last ? hf : nullptr,
            last ? cf : nullptr);

        lstm_step<0><<<32, 256, 0, stream>>>(
            (const void*)h0out, h1prev, W1, bih1, bhh1,
            C1, h1out,
            /*out_f32=*/out + (size_t)t * STEPEL,
            last ? (hf + STEPEL) : nullptr,
            last ? (cf + STEPEL) : nullptr);
    }
}
